// GptCovd_40913858461863
// MI455X (gfx1250) — matrix operand order repaired and checked by reading, not compiled, not run on hardware
//
#include <hip/hip_runtime.h>
#include <math.h>

// ---------------------------------------------------------------------------
// GPT-like decoder (LSTM + 6 transformer layers) for MI455X / gfx1250.
// All large GEMMs use v_wmma_f32_16x16x32_bf16 (bf16 in, f32 accumulate).
// - bf16 weights (~95MB) converted once per call -> resident in 192MB L2.
// - GEMM: block tile 128x64, wave tile 32x32 (4 WMMA per K-step, 2x frag reuse)
// - A tiles staged to LDS via the Tensor Data Mover (tensor_load_to_lds +
//   s_wait_tensorcnt), the CDNA5 path that frees VALU/VMEM for the XDL pipe.
// - Fused per-(b,h) attention: QK^T WMMA -> masked softmax in LDS -> PV WMMA.
//
// Input order assumption (setup_inputs() dict insertion order, nested dicts
// flattened in insertion order):
//  0 zeo[256,7] 1 syn[256,17] 2 smis_seq[256,125](int32)
//  3 W_zeo[7,768] 4 b_zeo 5 W_syn[17,768] 6 b_syn 7 char_emb[64,768]
//  8 type_emb[3,768] 9 Wih[3072,768] 10 Whh[3072,768] 11 bih 12 bhh
//  13 n1_g 14 n1_b 15 n2_g 16 n2_b 17 W_out[768,64] 18 b_out
//  19 ln1_g[6,768] 20 ln1_b 21 Wq[6,768,768] 22 bq 23 Wk 24 bk 25 Wv 26 bv
//  27 Wo 28 bo 29 ln2_g 30 ln2_b 31 W1[6,768,3072] 32 b1 33 W2[6,3072,768] 34 b2
// ---------------------------------------------------------------------------

typedef __attribute__((ext_vector_type(16))) __bf16 v16bf;
typedef __attribute__((ext_vector_type(8)))  float  v8f;
typedef __attribute__((ext_vector_type(4)))  unsigned int u32x4;
typedef __attribute__((ext_vector_type(8)))  int    i32x8;
typedef __attribute__((ext_vector_type(4)))  int    i32x4;

#if defined(__has_builtin)
#  if __has_builtin(__builtin_amdgcn_tensor_load_to_lds)
#    define HAVE_TDM 1
#  else
#    define HAVE_TDM 0
#  endif
#else
#  define HAVE_TDM 0
#endif

#define D_MODEL 768
#define SEQ_T   127
#define BATCH   256
#define MROWS   (BATCH * SEQ_T)   // 32512, multiple of 128
#define NHEAD   12
#define HDIM    64
#define NLAYER  6
#define FFDIM   3072
#define CHOUT   64

// ------------------------------ helpers -----------------------------------

__global__ void cvt_f32_bf16_kernel(const float* __restrict__ s,
                                    __bf16* __restrict__ d, int n) {
  int i = blockIdx.x * 256 + threadIdx.x;
  if (i < n) d[i] = (__bf16)s[i];
}

// src [R][C] f32 -> dst [C][R] bf16  (for Wih/Whh: x @ W.T needs W^T[K,N])
__global__ void cvt_transpose_kernel(const float* __restrict__ s,
                                     __bf16* __restrict__ d, int R, int C) {
  int i = blockIdx.x * 256 + threadIdx.x;
  if (i < R * C) {
    int r = i / C, c = i % C;
    d[(size_t)c * R + r] = (__bf16)s[i];
  }
}

__global__ void addvec_kernel(const float* __restrict__ a,
                              const float* __restrict__ b,
                              float* __restrict__ o, int n) {
  int i = blockIdx.x * 256 + threadIdx.x;
  if (i < n) o[i] = a[i] + b[i];
}

__global__ void zero_kernel(float* __restrict__ p, int n) {
  int i = blockIdx.x * 256 + threadIdx.x;
  if (i < n) p[i] = 0.0f;
}

// ------------------------------ embedding ---------------------------------

__global__ void embed_kernel(const float* __restrict__ zeo,
                             const float* __restrict__ syn,
                             const int*   __restrict__ smis,
                             const float* __restrict__ Wz, const float* __restrict__ bz,
                             const float* __restrict__ Ws, const float* __restrict__ bs,
                             const float* __restrict__ ce, const float* __restrict__ te,
                             float* __restrict__ x) {
  int idx = blockIdx.x * 256 + threadIdx.x;
  if (idx >= MROWS * D_MODEL) return;
  int d  = idx % D_MODEL;
  int bt = idx / D_MODEL;
  int t  = bt % SEQ_T;
  int b  = bt / SEQ_T;
  float v;
  if (t == 0) {
    v = bz[d] + te[d];
    #pragma unroll
    for (int i = 0; i < 7; ++i) v += zeo[b * 7 + i] * Wz[i * D_MODEL + d];
  } else if (t == 1) {
    v = bs[d] + te[D_MODEL + d];
    #pragma unroll
    for (int i = 0; i < 17; ++i) v += syn[b * 17 + i] * Ws[i * D_MODEL + d];
  } else {
    int pos = t - 2;
    int tok = smis[b * 125 + pos];
    float ang = (float)pos * __expf(-(float)d * (logf(10000.0f) / (float)D_MODEL));
    float pe  = (d & 1) ? __cosf(ang) : __sinf(ang);
    v = ce[(size_t)tok * D_MODEL + d] + pe + te[2 * D_MODEL + d];
  }
  x[idx] = v;
}

// ------------------------------ layernorm ---------------------------------

__global__ __launch_bounds__(256)
void ln_kernel(const float* __restrict__ x,
               const float* __restrict__ g, const float* __restrict__ bb,
               float* __restrict__ outf, __bf16* __restrict__ outb) {
  __shared__ float red[256];
  int row = blockIdx.x, tid = threadIdx.x;
  const float* xr = x + (size_t)row * D_MODEL;
  float v0 = xr[tid], v1 = xr[tid + 256], v2 = xr[tid + 512];
  red[tid] = v0 + v1 + v2;
  __syncthreads();
  for (int s = 128; s > 0; s >>= 1) {
    if (tid < s) red[tid] += red[tid + s];
    __syncthreads();
  }
  float mean = red[0] * (1.0f / D_MODEL);
  __syncthreads();
  float d0 = v0 - mean, d1 = v1 - mean, d2 = v2 - mean;
  red[tid] = d0 * d0 + d1 * d1 + d2 * d2;
  __syncthreads();
  for (int s = 128; s > 0; s >>= 1) {
    if (tid < s) red[tid] += red[tid + s];
    __syncthreads();
  }
  float rstd = rsqrtf(red[0] * (1.0f / D_MODEL) + 1e-5f);
  float o0 = d0 * rstd * g[tid] + bb[tid];
  float o1 = d1 * rstd * g[tid + 256] + bb[tid + 256];
  float o2 = d2 * rstd * g[tid + 512] + bb[tid + 512];
  if (outb) {
    __bf16* ob = outb + (size_t)row * D_MODEL;
    ob[tid] = (__bf16)o0; ob[tid + 256] = (__bf16)o1; ob[tid + 512] = (__bf16)o2;
  } else {
    float* of = outf + (size_t)row * D_MODEL;
    of[tid] = o0; of[tid + 256] = o1; of[tid + 512] = o2;
  }
}

// ------------------------------ WMMA GEMM ---------------------------------
// C[M,N] = A[M,K](bf16,row) @ B[K,N](bf16,row) [+bias] [+resid] [gelu]
// 256 threads = 8 waves; block tile 128x64; wave tile 32x32 (4 WMMA / K-step).
// A tile (128x32, row stride K) staged via Tensor Data Mover; B tile staged
// transposed ([n][k]) so B fragments are 128-bit LDS loads.

__device__ __forceinline__ float gelu_exact(float x) {
  return 0.5f * x * (1.0f + erff(x * 0.70710678118654752f));
}

__global__ __launch_bounds__(256)
void gemm_bf16_kernel(const __bf16* __restrict__ A,
                      const __bf16* __restrict__ Bw,
                      float* __restrict__ outf, __bf16* __restrict__ outb,
                      const float* __restrict__ bias,
                      const float* __restrict__ resid,
                      int M, int N, int K, int do_gelu) {
  __shared__ __bf16 a_lds[128][32];  // [m][k]
  __shared__ __bf16 bt_lds[64][32];  // [n][k] (transposed tile)

  int tid  = threadIdx.x;
  int wave = tid >> 5, lane = tid & 31;
  int half = lane >> 4, lrow = lane & 15;
  int m0 = blockIdx.y * 128, n0 = blockIdx.x * 64;
  int msub = (wave & 3) * 32;
  int nsub = (wave >> 2) * 32;

  v8f acc[2][2] = {};

#if HAVE_TDM
  unsigned lds_a = (unsigned)(uintptr_t)&a_lds[0][0];  // low 32b = LDS offset
#endif

  for (int kb = 0; kb < K; kb += 32) {
#if HAVE_TDM
    // ---- A tile via TDM: 2D tile 32(x) x 128(y), 2B elems, row stride K ----
    if (wave == 0) {
      unsigned long long ga =
          (unsigned long long)(uintptr_t)(A + (size_t)m0 * K + kb);
      u32x4 g0; i32x8 g1; i32x4 g2, g3; i32x8 g4;
      g0[0] = 1u;                                   // count=1, user descriptor
      g0[1] = lds_a;                                // lds_addr
      g0[2] = (unsigned)ga;                         // global_addr[31:0]
      g0[3] = (unsigned)(ga >> 32) | (2u << 30);    // addr[56:32] | type=2
      g1[0] = (int)(1u << 16);                      // data_size=1 -> 2 bytes
      g1[1] = (int)(((unsigned)K & 0xFFFFu) << 16); // tensor_dim0 lo16
      g1[2] = (int)(((unsigned)K >> 16) | (128u << 16)); // td0 hi | td1 lo (=128)
      g1[3] = (int)(32u << 16);                     // td1 hi=0 | tile_dim0=32
      g1[4] = (int)128;                             // tile_dim1=128, tile_dim2=0
      g1[5] = (int)(unsigned)K;                     // dim0_stride lo32 (elems)
      g1[6] = 0;                                    // stride hi bits
      g1[7] = 0;
      g2[0] = g2[1] = g2[2] = g2[3] = 0;
      g3[0] = g3[1] = g3[2] = g3[3] = 0;
      g4[0] = g4[1] = g4[2] = g4[3] = 0;
      g4[4] = g4[5] = g4[6] = g4[7] = 0;
      __builtin_amdgcn_tensor_load_to_lds(g0, g1, g2, g3, g4, 0);
    }
#else
    // ---- fallback: per-thread vector loads of A tile ----
    {
      int r = tid >> 1;                 // 0..127
      int c = (tid & 1) * 16;           // 0,16
      u32x4 v0 = *(const u32x4*)(A + (size_t)(m0 + r) * K + kb + c);
      u32x4 v1 = *(const u32x4*)(A + (size_t)(m0 + r) * K + kb + c + 8);
      *(u32x4*)&a_lds[r][c]     = v0;
      *(u32x4*)&a_lds[r][c + 8] = v1;
    }
#endif
    // ---- B tile: 32 rows x 64 cols, stored transposed into LDS ----
    {
      int kk = tid >> 3;
      int c  = (tid & 7) * 8;
      __bf16 tmp[8];
      *(u32x4*)tmp = *(const u32x4*)(Bw + (size_t)(kb + kk) * N + n0 + c);
      #pragma unroll
      for (int i = 0; i < 8; ++i) bt_lds[c + i][kk] = tmp[i];
    }
    if (kb + 32 < K)
      __builtin_prefetch(Bw + (size_t)(kb + 32 + (tid >> 3)) * N + n0, 0, 1);
#if HAVE_TDM
    if (wave == 0) __builtin_amdgcn_s_wait_tensorcnt(0);
#endif
    __syncthreads();

    // A fragments (16x32): lanes 0-15 K 0..15, lanes 16-31 K 16..31
    v16bf a0, a1;
    #pragma unroll
    for (int j = 0; j < 8; ++j) {
      int k0 = half * 16 + 2 * j;
      a0[2 * j]     = a_lds[msub + lrow][k0];
      a0[2 * j + 1] = a_lds[msub + lrow][k0 + 1];
      a1[2 * j]     = a_lds[msub + 16 + lrow][k0];
      a1[2 * j + 1] = a_lds[msub + 16 + lrow][k0 + 1];
    }
    // B fragments (32x16): lane-half selects K half, N = lane&15
    v16bf b0, b1;
    #pragma unroll
    for (int j = 0; j < 8; ++j) {
      int k0 = half * 16 + 2 * j;
      b0[2 * j]     = bt_lds[nsub + lrow][k0];
      b0[2 * j + 1] = bt_lds[nsub + lrow][k0 + 1];
      b1[2 * j]     = bt_lds[nsub + 16 + lrow][k0];
      b1[2 * j + 1] = bt_lds[nsub + 16 + lrow][k0 + 1];
    }
    acc[0][0] = __builtin_amdgcn_wmma_f32_16x16x32_bf16(false, a0, false, b0,
                                                        (short)0, acc[0][0], false, false);
    acc[0][1] = __builtin_amdgcn_wmma_f32_16x16x32_bf16(false, a0, false, b1,
                                                        (short)0, acc[0][1], false, false);
    acc[1][0] = __builtin_amdgcn_wmma_f32_16x16x32_bf16(false, a1, false, b0,
                                                        (short)0, acc[1][0], false, false);
    acc[1][1] = __builtin_amdgcn_wmma_f32_16x16x32_bf16(false, a1, false, b1,
                                                        (short)0, acc[1][1], false, false);
    __syncthreads();
  }

  // Epilogue. C layout: VGPR r -> M = r + 8*half, N = lane&15.
  #pragma unroll
  for (int im = 0; im < 2; ++im) {
    #pragma unroll
    for (int in_ = 0; in_ < 2; ++in_) {
      #pragma unroll
      for (int r = 0; r < 8; ++r) {
        int m = m0 + msub + im * 16 + half * 8 + r;
        int n = n0 + nsub + in_ * 16 + lrow;
        size_t o = (size_t)m * N + n;
        float t = acc[im][in_][r] + (bias ? bias[n] : 0.0f);
        if (resid) t += resid[o];
        if (do_gelu) t = gelu_exact(t);
        if (outb) outb[o] = (__bf16)t;
        else      outf[o] = t;
      }
    }
  }
}

// ------------------------------ LSTM pointwise ----------------------------

__global__ void lstm_point_kernel(const float* __restrict__ ghh,
                                  const __bf16* __restrict__ gih,
                                  float* __restrict__ c,
                                  __bf16* __restrict__ hb,
                                  float* __restrict__ x, int t) {
  int idx = blockIdx.x * 256 + threadIdx.x;
  if (idx >= BATCH * D_MODEL) return;
  int b = idx / D_MODEL, d = idx % D_MODEL;
  size_t gi_off = ((size_t)b * SEQ_T + t) * FFDIM + d;
  size_t gh_off = (size_t)b * FFDIM + d;
  float gi = (float)gih[gi_off]        + ghh[gh_off];
  float gf = (float)gih[gi_off + 768]  + ghh[gh_off + 768];
  float gg = (float)gih[gi_off + 1536] + ghh[gh_off + 1536];
  float go = (float)gih[gi_off + 2304] + ghh[gh_off + 2304];
  gi = 1.0f / (1.0f + __expf(-gi));
  gf = 1.0f / (1.0f + __expf(-gf));
  go = 1.0f / (1.0f + __expf(-go));
  gg = tanhf(gg);
  float cn = gf * c[idx] + gi * gg;
  float h  = go * tanhf(cn);
  c[idx]  = cn;
  hb[idx] = (__bf16)h;
  x[((size_t)b * SEQ_T + t) * D_MODEL + d] += h;  // residual: lstm(x) + x
}

// ------------------------------ fused attention ---------------------------
// One workgroup per (b,h): S = QK^T/8 (WMMA) -> mask -> softmax -> P@V (WMMA)
// T padded 127->128. Dynamic LDS ~145KB (<160KB => 2 WGs/WGP).

#define ATT_SMEM ((128*64 + 128*64 + 64*128) * 2 + 128*128*4 + 128*128*2 + 128*4)

__global__ __launch_bounds__(256)
void attn_kernel(const __bf16* __restrict__ qg,
                 const __bf16* __restrict__ kg,
                 const __bf16* __restrict__ vg,
                 const int* __restrict__ smis,
                 __bf16* __restrict__ og) {
  extern __shared__ char smem[];
  __bf16* q_lds  = (__bf16*)smem;               // [t][d]   128x64
  __bf16* kt_lds = q_lds + 128 * 64;            // [t][d]   128x64 (== B^T layout)
  __bf16* vt_lds = kt_lds + 128 * 64;           // [d][t]   64x128 (V transposed)
  float*  s_lds  = (float*)(vt_lds + 64 * 128); // scores   128x128 f32
  __bf16* p_lds  = (__bf16*)(s_lds + 128 * 128);// probs    128x128 bf16
  float*  msk    = (float*)(p_lds + 128 * 128); // per-key mask add (128)

  int tid = threadIdx.x;
  int bh = blockIdx.x;
  int h = bh % NHEAD, b = bh / NHEAD;

  if (tid < 128) {
    float mv = 0.0f;
    if (tid >= SEQ_T) mv = -1e30f;                                  // padded key
    else if (tid >= 2 && smis[b * 125 + tid - 2] == 0) mv = -1e30f; // '?' pad
    msk[tid] = mv;
  }
  #pragma unroll
  for (int it = 0; it < 4; ++it) {
    int idx = it * 2048 + tid * 8;
    int t = idx >> 6, d = idx & 63;
    __bf16 tq[8], tk[8], tv[8];
    if (t < SEQ_T) {
      size_t go_ = ((size_t)b * SEQ_T + t) * D_MODEL + h * HDIM + d;
      *(u32x4*)tq = *(const u32x4*)(qg + go_);
      *(u32x4*)tk = *(const u32x4*)(kg + go_);
      *(u32x4*)tv = *(const u32x4*)(vg + go_);
    } else {
      #pragma unroll
      for (int i = 0; i < 8; ++i) { tq[i] = (__bf16)0.0f; tk[i] = (__bf16)0.0f; tv[i] = (__bf16)0.0f; }
    }
    *(u32x4*)&q_lds[t * 64 + d]  = *(u32x4*)tq;
    *(u32x4*)&kt_lds[t * 64 + d] = *(u32x4*)tk;
    #pragma unroll
    for (int i = 0; i < 8; ++i) vt_lds[(d + i) * 128 + t] = tv[i];
  }
  __syncthreads();

  int wave = tid >> 5, lane = tid & 31, half = lane >> 4, lrow = lane & 15;

  // ---- S = Q K^T : 8x8 tiles of 16x16, one m-stripe per wave ----
  for (int i = 0; i < 8; ++i) {
    int tileid = wave * 8 + i;
    int mt = (tileid >> 3) * 16, nt = (tileid & 7) * 16;
    v8f acc = {};
    #pragma unroll
    for (int kb = 0; kb < 64; kb += 32) {
      v16bf af, bf_;
      #pragma unroll
      for (int j = 0; j < 8; ++j) {
        int ka = half * 16 + 2 * j + kb;
        af[2 * j]     = q_lds[(mt + lrow) * 64 + ka];
        af[2 * j + 1] = q_lds[(mt + lrow) * 64 + ka + 1];
        int kbn = half * 16 + 2 * j + kb;
        bf_[2 * j]     = kt_lds[(nt + lrow) * 64 + kbn];
        bf_[2 * j + 1] = kt_lds[(nt + lrow) * 64 + kbn + 1];
      }
      acc = __builtin_amdgcn_wmma_f32_16x16x32_bf16(false, af, false, bf_,
                                                    (short)0, acc, false, false);
    }
    #pragma unroll
    for (int r = 0; r < 8; ++r) {
      int m = mt + half * 8 + r, n = nt + lrow;
      float s = acc[r] * 0.125f + msk[n] + ((n > m) ? -1e30f : 0.0f);
      s_lds[m * 128 + n] = s;
    }
  }
  __syncthreads();

  // ---- softmax: wave w owns rows w*16 .. w*16+15 ----
  for (int r = 0; r < 16; ++r) {
    int m = wave * 16 + r;
    float v0 = s_lds[m * 128 + lane];
    float v1 = s_lds[m * 128 + lane + 32];
    float v2 = s_lds[m * 128 + lane + 64];
    float v3 = s_lds[m * 128 + lane + 96];
    float mx = fmaxf(fmaxf(v0, v1), fmaxf(v2, v3));
    #pragma unroll
    for (int o = 16; o >= 1; o >>= 1) mx = fmaxf(mx, __shfl_xor(mx, o, 32));
    float e0 = __expf(v0 - mx), e1 = __expf(v1 - mx);
    float e2 = __expf(v2 - mx), e3 = __expf(v3 - mx);
    float sm = e0 + e1 + e2 + e3;
    #pragma unroll
    for (int o = 16; o >= 1; o >>= 1) sm += __shfl_xor(sm, o, 32);
    float rinv = 1.0f / sm;
    p_lds[m * 128 + lane]      = (__bf16)(e0 * rinv);
    p_lds[m * 128 + lane + 32] = (__bf16)(e1 * rinv);
    p_lds[m * 128 + lane + 64] = (__bf16)(e2 * rinv);
    p_lds[m * 128 + lane + 96] = (__bf16)(e3 * rinv);
  }
  __syncthreads();

  // ---- O = P @ V : M=128, N=64, K=128 ----
  for (int i = 0; i < 4; ++i) {
    int tileid = wave * 4 + i;
    int mt = (tileid >> 2) * 16, nt = (tileid & 3) * 16;
    v8f acc = {};
    #pragma unroll
    for (int kb = 0; kb < 128; kb += 32) {
      v16bf af, bf_;
      #pragma unroll
      for (int j = 0; j < 8; ++j) {
        int ka = half * 16 + 2 * j + kb;
        af[2 * j]     = p_lds[(mt + lrow) * 128 + ka];
        af[2 * j + 1] = p_lds[(mt + lrow) * 128 + ka + 1];
        int kbn = half * 16 + 2 * j + kb;
        bf_[2 * j]     = vt_lds[(nt + lrow) * 128 + kbn];
        bf_[2 * j + 1] = vt_lds[(nt + lrow) * 128 + kbn + 1];
      }
      acc = __builtin_amdgcn_wmma_f32_16x16x32_bf16(false, af, false, bf_,
                                                    (short)0, acc, false, false);
    }
    #pragma unroll
    for (int r = 0; r < 8; ++r) {
      int m = mt + half * 8 + r, n = nt + lrow;
      if (m < SEQ_T)
        og[((size_t)b * SEQ_T + m) * D_MODEL + h * HDIM + n] = (__bf16)acc[r];
    }
  }
}

// ------------------------------ orchestration -----------------------------

extern "C" void kernel_launch(void* const* d_in, const int* in_sizes, int n_in,
                              void* d_out, int out_size, void* d_ws, size_t ws_size,
                              hipStream_t stream) {
  (void)in_sizes; (void)n_in; (void)out_size; (void)ws_size;

  const float* zeo   = (const float*)d_in[0];
  const float* syn   = (const float*)d_in[1];
  const int*   smis  = (const int*)d_in[2];
  const float* W_zeo = (const float*)d_in[3];
  const float* b_zeo = (const float*)d_in[4];
  const float* W_syn = (const float*)d_in[5];
  const float* b_syn = (const float*)d_in[6];
  const float* ch_e  = (const float*)d_in[7];
  const float* ty_e  = (const float*)d_in[8];
  const float* Wih   = (const float*)d_in[9];
  const float* Whh   = (const float*)d_in[10];
  const float* bih   = (const float*)d_in[11];
  const float* bhh   = (const float*)d_in[12];
  const float* n1_g  = (const float*)d_in[13];
  const float* n1_b  = (const float*)d_in[14];
  const float* n2_g  = (const float*)d_in[15];
  const float* n2_b  = (const float*)d_in[16];
  const float* W_out = (const float*)d_in[17];
  const float* b_out = (const float*)d_in[18];
  const float* ln1_g = (const float*)d_in[19];
  const float* ln1_b = (const float*)d_in[20];
  const float* Wq    = (const float*)d_in[21];
  const float* bq    = (const float*)d_in[22];
  const float* Wk    = (const float*)d_in[23];
  const float* bk    = (const float*)d_in[24];
  const float* Wv    = (const float*)d_in[25];
  const float* bv    = (const float*)d_in[26];
  const float* Wo    = (const float*)d_in[27];
  const float* bo    = (const float*)d_in[28];
  const float* ln2_g = (const float*)d_in[29];
  const float* ln2_b = (const float*)d_in[30];
  const float* W1    = (const float*)d_in[31];
  const float* b1    = (const float*)d_in[32];
  const float* W2    = (const float*)d_in[33];
  const float* b2    = (const float*)d_in[34];

  // ---- workspace carve-out ----
  char* ws = (char*)d_ws;
  size_t off = 0;
  auto carve = [&](size_t bytes) -> char* {
    char* p = ws + off;
    off += (bytes + 255) & ~(size_t)255;
    return p;
  };
  const size_t DD = (size_t)D_MODEL * D_MODEL;
  const size_t DF = (size_t)D_MODEL * FFDIM;
  __bf16* WihT   = (__bf16*)carve(DF * 2);
  __bf16* WhhT   = (__bf16*)carve(DF * 2);
  __bf16* WqB    = (__bf16*)carve(NLAYER * DD * 2);
  __bf16* WkB    = (__bf16*)carve(NLAYER * DD * 2);
  __bf16* WvB    = (__bf16*)carve(NLAYER * DD * 2);
  __bf16* WoB    = (__bf16*)carve(NLAYER * DD * 2);
  __bf16* W1B    = (__bf16*)carve(NLAYER * DF * 2);
  __bf16* W2B    = (__bf16*)carve(NLAYER * DF * 2);
  __bf16* WoutB  = (__bf16*)carve((size_t)D_MODEL * CHOUT * 2);
  float*  biasIH = (float*)carve(FFDIM * 4);
  float*  x      = (float*)carve((size_t)MROWS * D_MODEL * 4);   // residual stream
  __bf16* xb     = (__bf16*)carve((size_t)MROWS * D_MODEL * 2);  // LN output (bf16)
  __bf16* qb     = (__bf16*)carve((size_t)MROWS * D_MODEL * 2);
  __bf16* kb2    = (__bf16*)carve((size_t)MROWS * D_MODEL * 2);
  __bf16* vb2    = (__bf16*)carve((size_t)MROWS * D_MODEL * 2);
  __bf16* atnb   = (__bf16*)carve((size_t)MROWS * D_MODEL * 2);
  __bf16* gbig   = (__bf16*)carve((size_t)MROWS * FFDIM * 2);    // gates_ih / gelu(h)
  float*  ghh    = (float*)carve((size_t)BATCH * FFDIM * 4);
  __bf16* hB     = (__bf16*)carve((size_t)BATCH * D_MODEL * 2);
  float*  cst    = (float*)carve((size_t)BATCH * D_MODEL * 4);

  auto blocks = [](size_t n) { return (unsigned)((n + 255) / 256); };
  auto gemm = [&](const __bf16* A, const __bf16* Bw, float* outf, __bf16* outb,
                  const float* bias, const float* resid, int M, int N, int K,
                  int gelu) {
    dim3 grid(N / 64, M / 128);
    gemm_bf16_kernel<<<grid, 256, 0, stream>>>(A, Bw, outf, outb, bias, resid,
                                               M, N, K, gelu);
  };

  // ---- weight conversion (bf16 weights ~95MB -> resident in 192MB L2) ----
  cvt_transpose_kernel<<<blocks(DF), 256, 0, stream>>>(Wih, WihT, FFDIM, D_MODEL);
  cvt_transpose_kernel<<<blocks(DF), 256, 0, stream>>>(Whh, WhhT, FFDIM, D_MODEL);
  cvt_f32_bf16_kernel<<<blocks(NLAYER * DD), 256, 0, stream>>>(Wq, WqB, NLAYER * DD);
  cvt_f32_bf16_kernel<<<blocks(NLAYER * DD), 256, 0, stream>>>(Wk, WkB, NLAYER * DD);
  cvt_f32_bf16_kernel<<<blocks(NLAYER * DD), 256, 0, stream>>>(Wv, WvB, NLAYER * DD);
  cvt_f32_bf16_kernel<<<blocks(NLAYER * DD), 256, 0, stream>>>(Wo, WoB, NLAYER * DD);
  cvt_f32_bf16_kernel<<<blocks(NLAYER * DF), 256, 0, stream>>>(W1, W1B, NLAYER * DF);
  cvt_f32_bf16_kernel<<<blocks(NLAYER * DF), 256, 0, stream>>>(W2, W2B, NLAYER * DF);
  cvt_f32_bf16_kernel<<<blocks((size_t)D_MODEL * CHOUT), 256, 0, stream>>>(
      W_out, WoutB, D_MODEL * CHOUT);
  addvec_kernel<<<blocks(FFDIM), 256, 0, stream>>>(bih, bhh, biasIH, FFDIM);

  // ---- embeddings ----
  embed_kernel<<<blocks((size_t)MROWS * D_MODEL), 256, 0, stream>>>(
      zeo, syn, smis, W_zeo, b_zeo, W_syn, b_syn, ch_e, ty_e, x);
  cvt_f32_bf16_kernel<<<blocks((size_t)MROWS * D_MODEL), 256, 0, stream>>>(
      x, xb, MROWS * D_MODEL);

  // ---- LSTM: input-gate GEMM for all timesteps at once ----
  gemm(xb, WihT, nullptr, gbig, biasIH, nullptr, MROWS, FFDIM, D_MODEL, 0);

  zero_kernel<<<blocks((size_t)BATCH * D_MODEL / 2), 256, 0, stream>>>(
      (float*)hB, BATCH * D_MODEL / 2);
  zero_kernel<<<blocks((size_t)BATCH * D_MODEL), 256, 0, stream>>>(
      cst, BATCH * D_MODEL);

  // ---- LSTM recurrence: 127 dependent (GEMM + pointwise) steps ----
  for (int t = 0; t < SEQ_T; ++t) {
    gemm(hB, WhhT, ghh, nullptr, nullptr, nullptr, BATCH, FFDIM, D_MODEL, 0);
    lstm_point_kernel<<<blocks((size_t)BATCH * D_MODEL), 256, 0, stream>>>(
        ghh, gbig, cst, hB, x, t);
  }

  // ---- post-LSTM layernorm (in place) ----
  ln_kernel<<<MROWS, 256, 0, stream>>>(x, n1_g, n1_b, x, nullptr);

  // ---- 6 decoder layers ----
  for (int l = 0; l < NLAYER; ++l) {
    const __bf16* wq = WqB + (size_t)l * DD;
    const __bf16* wk = WkB + (size_t)l * DD;
    const __bf16* wv = WvB + (size_t)l * DD;
    const __bf16* wo = WoB + (size_t)l * DD;
    const __bf16* w1 = W1B + (size_t)l * DF;
    const __bf16* w2 = W2B + (size_t)l * DF;

    ln_kernel<<<MROWS, 256, 0, stream>>>(x, ln1_g + l * D_MODEL,
                                         ln1_b + l * D_MODEL, nullptr, xb);
    gemm(xb, wq, nullptr, qb, bq + l * D_MODEL, nullptr, MROWS, D_MODEL, D_MODEL, 0);
    gemm(xb, wk, nullptr, kb2, bk + l * D_MODEL, nullptr, MROWS, D_MODEL, D_MODEL, 0);
    gemm(xb, wv, nullptr, vb2, bv + l * D_MODEL, nullptr, MROWS, D_MODEL, D_MODEL, 0);

    attn_kernel<<<BATCH * NHEAD, 256, ATT_SMEM, stream>>>(qb, kb2, vb2, smis, atnb);

    gemm(atnb, wo, x, nullptr, bo + l * D_MODEL, x, MROWS, D_MODEL, D_MODEL, 0);

    ln_kernel<<<MROWS, 256, 0, stream>>>(x, ln2_g + l * D_MODEL,
                                         ln2_b + l * D_MODEL, nullptr, xb);
    gemm(xb, w1, nullptr, gbig, b1 + l * FFDIM, nullptr, MROWS, FFDIM, D_MODEL, 1);
    gemm(gbig, w2, x, nullptr, b2 + l * D_MODEL, x, MROWS, D_MODEL, FFDIM, 0);
  }

  // ---- final LN + output projection into d_out (f32 [32512, 64]) ----
  ln_kernel<<<MROWS, 256, 0, stream>>>(x, n2_g, n2_b, nullptr, xb);
  gemm(xb, WoutB, (float*)d_out, nullptr, b_out, nullptr, MROWS, CHOUT, D_MODEL, 0);
}
